// SoftAttention_15779709846002
// MI455X (gfx1250) — matrix operand order repaired and checked by reading, not compiled, not run on hardware
//
#include <hip/hip_runtime.h>
#include <hip/hip_bf16.h>
#include <math.h>
#include <stdint.h>

typedef __bf16 bf16;
typedef __attribute__((ext_vector_type(16))) __bf16 v16bf;
typedef __attribute__((ext_vector_type(8)))  __bf16 v8bf;
typedef __attribute__((ext_vector_type(8)))  float  v8f;

constexpr int NB = 32;    // batch
constexpr int LL = 512;   // sequence length
constexpr int DD = 768;   // feature dim
constexpr int DC = 3072;  // 4*DD combined dim

// GEMM tiling
constexpr int BM = 128, BN = 128, BK = 32;
constexpr int LDK = 40;   // padded K-stride in LDS (bf16 elems) to dodge bank conflicts

// ---------------------------------------------------------------------------
// CDNA5 async global->LDS copy (ASYNCcnt path, no VGPR round-trip).
// GVS mode: mem_addr = SGPR64 base + per-lane signed 32-bit byte offset.
// LDS destination = low 32 bits of generic shared pointer (wave LDS offset).
// ---------------------------------------------------------------------------
__device__ __forceinline__ void async_copy_b128(const bf16* lds_dst, uint32_t gbyte_off,
                                                const bf16* gbase) {
    const uint32_t lds_off = (uint32_t)(uintptr_t)lds_dst;
    const uint64_t sbase = (uint64_t)(uintptr_t)gbase;
    asm volatile("global_load_async_to_lds_b128 %0, %1, %2"
                 :: "v"(lds_off), "v"(gbyte_off), "s"(sbase)
                 : "memory");
}
__device__ __forceinline__ void wait_async0() {
    asm volatile("s_wait_asynccnt 0" ::: "memory");
}

// ---------------------------------------------------------------------------
// WMMA fragment loaders (ISA 7.12.2 layouts, wave32)
// A 16x32 bf16: lanes 0-15 = M rows, half=0 -> K 0..7 & 16..23, half=1 -> K 8..15 & 24..31
// B 32x16 bf16: lanes 0-15 = N cols, half=0 -> K 0..7 & 16..23, half=1 -> K 8..15 & 24..31
// Tiles stored row-major with row stride LDK; B tile stored [n][k].
// ---------------------------------------------------------------------------
__device__ __forceinline__ v16bf make16(v8bf lo, v8bf hi) {
    v16bf r;
#pragma unroll
    for (int t = 0; t < 8; ++t) { r[t] = lo[t]; r[t + 8] = hi[t]; }
    return r;
}

__device__ __forceinline__ v16bf fragA(const bf16* base, int lrow, int half) {
    const bf16* p = base + lrow * LDK + half * 8;
    return make16(*(const v8bf*)p, *(const v8bf*)(p + 16));
}

__device__ __forceinline__ v16bf fragB(const bf16* base, int lrow, int half) {
    const bf16* p = base + lrow * LDK + half * 8;
    return make16(*(const v8bf*)p, *(const v8bf*)(p + 16));
}

// ---------------------------------------------------------------------------
// Kernel 1: fp32 -> bf16 convert, producing row-major copy and [d][l] transpose
// grid (DD/32, LL/32, NB), block 256
// ---------------------------------------------------------------------------
__global__ __launch_bounds__(256) void k_cvt(const float* __restrict__ q,
                                             bf16* __restrict__ qbf,
                                             bf16* __restrict__ qbfT) {
    __shared__ bf16 s[32][33];
    const int b = blockIdx.z, i0 = blockIdx.y * 32, d0 = blockIdx.x * 32;
    const int x = threadIdx.x & 31, y0 = threadIdx.x >> 5;
    const float* src = q + ((size_t)b * LL + i0) * DD + d0;
    bf16* dst = qbf + ((size_t)b * LL + i0) * DD + d0;
#pragma unroll
    for (int r = 0; r < 4; ++r) {
        const int row = y0 + r * 8;
        const float v = src[(size_t)row * DD + x];
        const bf16 h = (bf16)v;
        dst[(size_t)row * DD + x] = h;
        s[row][x] = h;
    }
    __syncthreads();
    bf16* dstT = qbfT + ((size_t)b * DD + d0) * LL + i0;
#pragma unroll
    for (int r = 0; r < 4; ++r) {
        const int row = y0 + r * 8;
        dstT[(size_t)row * LL + x] = s[x][row];
    }
}

// ---------------------------------------------------------------------------
// Kernel 2: att[b] = q1[b] (LLxDD) * q2[b]^T  -> [LL x LL] fp32
// grid (LL/BN, LL/BM, NB), block 256 (8 waves; wave grid 2(M) x 4(N); 64x32/wave)
// Staging uses async global->LDS b128 copies (ASYNCcnt).
// ---------------------------------------------------------------------------
__global__ __launch_bounds__(256) void k_att(const bf16* __restrict__ q1bf,
                                             const bf16* __restrict__ q2bf,
                                             float* __restrict__ att) {
    __shared__ bf16 As[2][BM * LDK];
    __shared__ bf16 Bs[2][BN * LDK];
    const int b = blockIdx.z, m0 = blockIdx.y * BM, n0 = blockIdx.x * BN;
    const bf16* Ag = q1bf + ((size_t)b * LL + m0) * DD;
    const bf16* Bg = q2bf + ((size_t)b * LL + n0) * DD;
    const int tid = threadIdx.x;
    const int wid = tid >> 5, lane = tid & 31;
    const int wm = wid & 1, wn = wid >> 1;
    const int half = lane >> 4, lrow = lane & 15;

    v8f acc[4][2];
#pragma unroll
    for (int mi = 0; mi < 4; ++mi)
#pragma unroll
        for (int ni = 0; ni < 2; ++ni) { v8f z = {}; acc[mi][ni] = z; }

    auto stage = [&](int buf, int k) {
#pragma unroll
        for (int c = 0; c < 2; ++c) {
            const int idx = tid + c * 256;
            const int row = idx >> 2, col = (idx & 3) * 8;
            const uint32_t goff = (uint32_t)(row * DD + k + col) * 2u;
            async_copy_b128(&As[buf][row * LDK + col], goff, Ag);
            async_copy_b128(&Bs[buf][row * LDK + col], goff, Bg);
        }
    };

    stage(0, 0);
    wait_async0();
    __syncthreads();
    int buf = 0;
    constexpr int KT = DD / BK;
    for (int kt = 0; kt < KT; ++kt) {
        if (kt + 1 < KT) stage(buf ^ 1, (kt + 1) * BK);
        v16bf a[4], bb[2];
#pragma unroll
        for (int mi = 0; mi < 4; ++mi) a[mi] = fragA(&As[buf][(wm * 64 + mi * 16) * LDK], lrow, half);
#pragma unroll
        for (int ni = 0; ni < 2; ++ni) bb[ni] = fragB(&Bs[buf][(wn * 32 + ni * 16) * LDK], lrow, half);
#pragma unroll
        for (int mi = 0; mi < 4; ++mi)
#pragma unroll
            for (int ni = 0; ni < 2; ++ni)
                acc[mi][ni] = __builtin_amdgcn_wmma_f32_16x16x32_bf16(
                    false, a[mi], false, bb[ni], (short)0, acc[mi][ni], false, false);
        wait_async0();
        __syncthreads();
        buf ^= 1;
    }

#pragma unroll
    for (int mi = 0; mi < 4; ++mi)
#pragma unroll
        for (int ni = 0; ni < 2; ++ni)
#pragma unroll
            for (int r = 0; r < 8; ++r) {
                const int m = m0 + wm * 64 + mi * 16 + half * 8 + r;
                const int n = n0 + wn * 32 + ni * 16 + lrow;
                att[((size_t)b * LL + m) * LL + n] = acc[mi][ni][r];
            }
}

// ---------------------------------------------------------------------------
// Kernel 3: per-row (over j) softmax stats: rm/rs indexed [b*LL + i]
// one wave per row; grid NB*LL/8 blocks, 256 threads
// ---------------------------------------------------------------------------
__global__ __launch_bounds__(256) void k_rowstats(const float* __restrict__ att,
                                                  float* __restrict__ rm,
                                                  float* __restrict__ rs) {
    const int wid = threadIdx.x >> 5, lane = threadIdx.x & 31;
    const int row = blockIdx.x * 8 + wid;
    const float* p = att + (size_t)row * LL;
    float v[16];
    float m = -INFINITY;
#pragma unroll
    for (int t = 0; t < 16; ++t) { v[t] = p[lane + t * 32]; m = fmaxf(m, v[t]); }
#pragma unroll
    for (int o = 16; o >= 1; o >>= 1) m = fmaxf(m, __shfl_xor(m, o, 32));
    float s = 0.f;
#pragma unroll
    for (int t = 0; t < 16; ++t) s += __expf(v[t] - m);
#pragma unroll
    for (int o = 16; o >= 1; o >>= 1) s += __shfl_xor(s, o, 32);
    if (lane == 0) { rm[row] = m; rs[row] = s; }
}

// ---------------------------------------------------------------------------
// Kernel 4: per-column (over i) softmax stats: cm/cs indexed [b*LL + j]
// one thread per column, coalesced column walk (att is L2 resident)
// grid NB*2 blocks, 256 threads
// ---------------------------------------------------------------------------
__global__ __launch_bounds__(256) void k_colstats(const float* __restrict__ att,
                                                  float* __restrict__ cm,
                                                  float* __restrict__ cs) {
    const int b = blockIdx.x >> 1;
    const int j = ((blockIdx.x & 1) << 8) + threadIdx.x;
    const float* p = att + (size_t)b * LL * LL + j;
    float m = -INFINITY;
    for (int i = 0; i < LL; ++i) m = fmaxf(m, p[(size_t)i * LL]);
    float s = 0.f;
    for (int i = 0; i < LL; ++i) s += __expf(p[(size_t)i * LL] - m);
    cm[b * LL + j] = m;
    cs[b * LL + j] = s;
}

// ---------------------------------------------------------------------------
// Kernel 5 (templated): aligned = softmax-weighted GEMM, fused concat output.
// TRANSA=true : out rows j; A[j][i] = exp(att[i][j]-cm[j]); B = q1 (q1bfT);
//               writes q2_combined = [q2, a, q2-a, q2*a]
// TRANSA=false: out rows i; A[i][j] = exp(att[i][j]-rm[i]); B = q2 (q2bfT);
//               writes q1_combined = [q1, a, q1-a, q1*a]
// A-tile staged through VALU (exp + bf16 cvt); B-tile staged async to LDS.
// grid (DD/BN, LL/BM, NB), block 256
// ---------------------------------------------------------------------------
template <bool TRANSA>
__global__ __launch_bounds__(256) void k_align(const float* __restrict__ att,
                                               const float* __restrict__ stM,
                                               const float* __restrict__ stS,
                                               const bf16* __restrict__ Bt,
                                               const float* __restrict__ X,
                                               float* __restrict__ outbase) {
    __shared__ bf16 As[2][BM * LDK];
    __shared__ bf16 Bs[2][BN * LDK];
    __shared__ float mS[BM];
    __shared__ float rS[BM];

    const int b = blockIdx.z, m0 = blockIdx.y * BM, n0 = blockIdx.x * BN;
    const int tid = threadIdx.x;
    const int wid = tid >> 5, lane = tid & 31;
    const int wm = wid & 1, wn = wid >> 1;
    const int half = lane >> 4, lrow = lane & 15;
    const bf16* BgT = Bt + (size_t)b * DD * LL;   // [d][l] layout

    if (tid < BM) {
        mS[tid] = stM[b * LL + m0 + tid];
        rS[tid] = 1.0f / stS[b * LL + m0 + tid];
    }
    __syncthreads();

    v8f acc[4][2];
#pragma unroll
    for (int mi = 0; mi < 4; ++mi)
#pragma unroll
        for (int ni = 0; ni < 2; ++ni) { v8f z = {}; acc[mi][ni] = z; }

    auto stageA = [&](int buf, int k) {
        if (TRANSA) {
            // read att[k+kr][m0+m4..+3], scatter transposed into As[m][kr]
#pragma unroll
            for (int c = 0; c < 4; ++c) {
                const int idx = tid + c * 256;
                const int kr = idx >> 5, m4 = (idx & 31) * 4;
                const float4 v = *(const float4*)&att[((size_t)b * LL + k + kr) * LL + m0 + m4];
                As[buf][(m4 + 0) * LDK + kr] = (bf16)__expf(v.x - mS[m4 + 0]);
                As[buf][(m4 + 1) * LDK + kr] = (bf16)__expf(v.y - mS[m4 + 1]);
                As[buf][(m4 + 2) * LDK + kr] = (bf16)__expf(v.z - mS[m4 + 2]);
                As[buf][(m4 + 3) * LDK + kr] = (bf16)__expf(v.w - mS[m4 + 3]);
            }
        } else {
#pragma unroll
            for (int c = 0; c < 4; ++c) {
                const int idx = tid + c * 256;
                const int mr = idx >> 3, k4 = (idx & 7) * 4;
                const float4 v = *(const float4*)&att[((size_t)b * LL + m0 + mr) * LL + k + k4];
                const float mm = mS[mr];
                As[buf][mr * LDK + k4 + 0] = (bf16)__expf(v.x - mm);
                As[buf][mr * LDK + k4 + 1] = (bf16)__expf(v.y - mm);
                As[buf][mr * LDK + k4 + 2] = (bf16)__expf(v.z - mm);
                As[buf][mr * LDK + k4 + 3] = (bf16)__expf(v.w - mm);
            }
        }
    };
    auto stageB = [&](int buf, int k) {
#pragma unroll
        for (int c = 0; c < 2; ++c) {
            const int idx = tid + c * 256;
            const int row = idx >> 2, col = (idx & 3) * 8;
            const uint32_t goff = (uint32_t)((n0 + row) * LL + k + col) * 2u;
            async_copy_b128(&Bs[buf][row * LDK + col], goff, BgT);
        }
    };

    stageA(0, 0);
    stageB(0, 0);
    wait_async0();
    __syncthreads();
    int buf = 0;
    constexpr int KT = LL / BK;
    for (int kt = 0; kt < KT; ++kt) {
        if (kt + 1 < KT) { stageA(buf ^ 1, (kt + 1) * BK); stageB(buf ^ 1, (kt + 1) * BK); }
        v16bf a[4], bb[2];
#pragma unroll
        for (int mi = 0; mi < 4; ++mi) a[mi] = fragA(&As[buf][(wm * 64 + mi * 16) * LDK], lrow, half);
#pragma unroll
        for (int ni = 0; ni < 2; ++ni) bb[ni] = fragB(&Bs[buf][(wn * 32 + ni * 16) * LDK], lrow, half);
#pragma unroll
        for (int mi = 0; mi < 4; ++mi)
#pragma unroll
            for (int ni = 0; ni < 2; ++ni)
                acc[mi][ni] = __builtin_amdgcn_wmma_f32_16x16x32_bf16(
                    false, a[mi], false, bb[ni], (short)0, acc[mi][ni], false, false);
        wait_async0();
        __syncthreads();
        buf ^= 1;
    }

    // epilogue: scale by 1/sum, fuse 4-way concat with X
#pragma unroll
    for (int mi = 0; mi < 4; ++mi)
#pragma unroll
        for (int ni = 0; ni < 2; ++ni)
#pragma unroll
            for (int r = 0; r < 8; ++r) {
                const int ml = wm * 64 + mi * 16 + half * 8 + r;
                const int m = m0 + ml;
                const int n = n0 + wn * 32 + ni * 16 + lrow;
                const float v = acc[mi][ni][r] * rS[ml];
                const float x = X[((size_t)b * LL + m) * DD + n];
                float* o = outbase + ((size_t)b * LL + m) * DC;
                o[n] = x;
                o[DD + n] = v;
                o[2 * DD + n] = x - v;
                o[3 * DD + n] = x * v;
            }
}

// ---------------------------------------------------------------------------
extern "C" void kernel_launch(void* const* d_in, const int* in_sizes, int n_in,
                              void* d_out, int out_size, void* d_ws, size_t ws_size,
                              hipStream_t stream) {
    const float* q1 = (const float*)d_in[0];
    const float* q2 = (const float*)d_in[1];
    float* out = (float*)d_out;

    const size_t QE = (size_t)NB * LL * DD;      // 12,582,912 elems
    const size_t AE = (size_t)NB * LL * LL;      // 8,388,608 elems

    bf16* q1bf  = (bf16*)d_ws;
    bf16* q2bf  = q1bf + QE;
    bf16* q1bfT = q2bf + QE;
    bf16* q2bfT = q1bfT + QE;
    float* att  = (float*)(q2bfT + QE);
    float* rm   = att + AE;
    float* rs   = rm + (size_t)NB * LL;
    float* cm   = rs + (size_t)NB * LL;
    float* cs   = cm + (size_t)NB * LL;

    float* q1c = out;                            // q1_combined [NB, LL, DC]
    float* q2c = out + (size_t)NB * LL * DC;     // q2_combined [NB, LL, DC]

    // 1. fp32 -> bf16 (+ transposed copies)
    k_cvt<<<dim3(DD / 32, LL / 32, NB), 256, 0, stream>>>(q1, q1bf, q1bfT);
    k_cvt<<<dim3(DD / 32, LL / 32, NB), 256, 0, stream>>>(q2, q2bf, q2bfT);

    // 2. att = q1 @ q2^T  (bf16 WMMA, f32 accumulate, async LDS staging)
    k_att<<<dim3(LL / BN, LL / BM, NB), 256, 0, stream>>>(q1bf, q2bf, att);

    // 3. softmax stats (att is L2-resident: 33.5 MB << 192 MB)
    k_rowstats<<<NB * LL / 8, 256, 0, stream>>>(att, rm, rs);
    k_colstats<<<NB * 2, 256, 0, stream>>>(att, cm, cs);

    // 4. q1_aligned = w1^T @ q1, fused into q2_combined
    k_align<true><<<dim3(DD / BN, LL / BM, NB), 256, 0, stream>>>(att, cm, cs, q1bfT, q2, q2c);
    // 5. q2_aligned = w2 @ q2, fused into q1_combined
    k_align<false><<<dim3(DD / BN, LL / BM, NB), 256, 0, stream>>>(att, rm, rs, q2bfT, q1, q1c);
}
